// CBOWModel_30288109372156
// MI455X (gfx1250) — matrix operand order repaired and checked by reading, not compiled, not run on hardware
//
#include <hip/hip_runtime.h>

typedef __attribute__((ext_vector_type(16))) _Float16 v16h;
typedef __attribute__((ext_vector_type(8)))  _Float16 v8h;
typedef __attribute__((ext_vector_type(4)))  _Float16 v4h;
typedef __attribute__((ext_vector_type(8)))  float    v8f;
typedef __attribute__((ext_vector_type(4)))  float    v4f;

#define DDIM 128
#define CCTX 8
#define NNEG 40          // C*K negatives per b
#define WPB  8           // waves per block (256 threads, 8 wave32)
#define ROWSTRIDE 136    // halves per staged row: 272B -> 4-bank shift/row, conflict-free

__device__ __forceinline__ float softplus_f(float x) {
    // log(1+exp(x)); x is clipped to [-10,10] so direct form is safe
    return __logf(1.0f + __expf(x));
}

__global__ __launch_bounds__(256) void cbow_loss_kernel(
    const float* __restrict__ u_weights,
    const float* __restrict__ v_weights,
    const int*   __restrict__ pos_u,
    const int*   __restrict__ pos_v,
    const int*   __restrict__ neg_v,
    float*       __restrict__ block_partials)
{
    __shared__ _Float16 vbuf[WPB][DDIM];               // context vector V (f16)
    __shared__ _Float16 rowbuf[WPB][16 * ROWSTRIDE];   // 16 staged rows per wave
    __shared__ float    wpart[WPB];

    const int tid  = threadIdx.x;
    const int w    = tid >> 5;
    const int lane = tid & 31;
    const int b    = blockIdx.x * WPB + w;
    const int grp  = (lane >= 16) ? 1 : 0;

    // ---- V = sum_c v_weights[pos_v[b*8+c]]; lane owns D-elems [4*lane,4*lane+4) ----
    v4f vacc = {0.0f, 0.0f, 0.0f, 0.0f};
    #pragma unroll
    for (int c = 0; c < CCTX; ++c) {
        int idx = pos_v[b * CCTX + c];
        const v4f* row = (const v4f*)(v_weights + (size_t)idx * DDIM);
        vacc += row[lane];                         // coalesced: wave reads full 512B row
    }
    {
        v4h h;
        h[0]=(_Float16)vacc[0]; h[1]=(_Float16)vacc[1];
        h[2]=(_Float16)vacc[2]; h[3]=(_Float16)vacc[3];
        *(v4h*)(&vbuf[w][lane * 4]) = h;
    }

    const int upos = pos_u[b * CCTX];
    float lossAcc = 0.0f;

    for (int t = 0; t < 3; ++t) {
        __syncthreads();
        // ---- stage 16 rows (f32 gather -> f16 LDS); row 0 overall is the U row ----
        for (int r = 0; r < 16; ++r) {
            int g = t * 16 + r;
            const float* base;
            if (g == 0) {
                base = u_weights + (size_t)upos * DDIM;
            } else {
                int n = (g <= NNEG) ? (g - 1) : (NNEG - 1);   // pad rows 41..47 (masked later)
                base = v_weights + (size_t)neg_v[b * NNEG + n] * DDIM;
            }
            v4f d = ((const v4f*)base)[lane];
            v4h h;
            h[0]=(_Float16)d[0]; h[1]=(_Float16)d[1];
            h[2]=(_Float16)d[2]; h[3]=(_Float16)d[3];
            *(v4h*)(&rowbuf[w][r * ROWSTRIDE + lane * 4]) = h;
        }
        __syncthreads();

        const _Float16* myrow = &rowbuf[w][(lane & 15) * ROWSTRIDE];
        const _Float16* vb    = &vbuf[w][0];

        v8f acc = {};
        #pragma unroll
        for (int k0 = 0; k0 < DDIM; k0 += 32) {
            // A 16x32 f16 layout: lanes 0-15 hold K k0+0..7 / k0+16..23 of row M=lane;
            //                     lanes 16-31 hold K k0+8..15 / k0+24..31 of row M=lane-16
            union { v16h v; v8h h[2]; } A, Bm;
            A.h[0]  = *(const v8h*)(myrow + k0 + grp * 8);
            A.h[1]  = *(const v8h*)(myrow + k0 + 16 + grp * 8);
            // B 32x16 with every column = V chunk: lane needs halves k0+grp*8..+8 / k0+16+grp*8..+8
            Bm.h[0] = *(const v8h*)(vb + k0 + grp * 8);
            Bm.h[1] = *(const v8h*)(vb + k0 + 16 + grp * 8);
            acc = __builtin_amdgcn_wmma_f32_16x16x32_f16(
                      false, A.v, false, Bm.v, (short)0, acc, false, false);
        }

        // All 16 columns of D are identical; lane 0 holds rows 0-7, lane 16 rows 8-15.
        if ((lane & 15) == 0) {
            #pragma unroll
            for (int j = 0; j < 8; ++j) {
                int g = t * 16 + j + grp * 8;
                float s = fminf(fmaxf(acc[j], -10.0f), 10.0f);
                if (g == 0)          lossAcc += softplus_f(-s);                      // pos loss
                else if (g <= NNEG)  lossAcc += softplus_f(s) * (1.0f / (float)NNEG); // neg loss
            }
        }
    }

    // combine lanes 0 and 16 -> per-b loss on lane 0
    lossAcc += __shfl_xor(lossAcc, 16, 32);
    if (lane == 0) wpart[w] = lossAcc;
    __syncthreads();
    if (tid == 0) {
        float s = 0.0f;
        #pragma unroll
        for (int i = 0; i < WPB; ++i) s += wpart[i];
        block_partials[blockIdx.x] = s;
    }
}

__global__ __launch_bounds__(256) void cbow_reduce_kernel(
    const float* __restrict__ partials, float* __restrict__ out,
    int n, float scale)
{
    __shared__ float sm[256];
    float s = 0.0f;
    for (int i = threadIdx.x; i < n; i += 256) s += partials[i];
    sm[threadIdx.x] = s;
    __syncthreads();
    for (int off = 128; off > 0; off >>= 1) {
        if ((int)threadIdx.x < off) sm[threadIdx.x] += sm[threadIdx.x + off];
        __syncthreads();
    }
    if (threadIdx.x == 0) out[0] = sm[0] * scale;
}

extern "C" void kernel_launch(void* const* d_in, const int* in_sizes, int n_in,
                              void* d_out, int out_size, void* d_ws, size_t ws_size,
                              hipStream_t stream) {
    const float* u_weights = (const float*)d_in[0];
    const float* v_weights = (const float*)d_in[1];
    const int*   pos_u     = (const int*)d_in[2];
    const int*   pos_v     = (const int*)d_in[3];
    const int*   neg_v     = (const int*)d_in[4];

    const int Bn      = in_sizes[2] / CCTX;   // pos_u has B*C entries
    const int nblocks = Bn / WPB;             // one wave per b, 8 waves/block
    float* partials   = (float*)d_ws;

    cbow_loss_kernel<<<nblocks, 256, 0, stream>>>(
        u_weights, v_weights, pos_u, pos_v, neg_v, partials);
    cbow_reduce_kernel<<<1, 256, 0, stream>>>(
        partials, (float*)d_out, nblocks, 1.0f / (float)Bn);
}
